// PVConvKerasModel_5471788335254
// MI455X (gfx1250) — matrix operand order repaired and checked by reading, not compiled, not run on hardware
//
#include <hip/hip_runtime.h>
#include <hip/hip_bf16.h>
#include <math.h>

// ---------------------------------------------------------------------------
// PVCNN (PVConv x6 + classifier) for MI455X / gfx1250, wave32 + WMMA.
// All GEMM operands are pre-packed to f16 with K padded to 32 so the WMMA
// inner loops contain only unconditional, aligned vector loads:
//   A fragment: 1 x 32B (v16h) load, B fragment: 1 x 32B (v16h) load, 1 WMMA.
// Conv boundary handling uses a zero "halo row" (voxel index NVOX) instead of
// per-element predication. f32 accumulate, fused bias+ReLU+BN epilogues.
// ---------------------------------------------------------------------------

typedef __attribute__((ext_vector_type(16))) _Float16 v16h;
typedef __attribute__((ext_vector_type(8)))  _Float16 v8h;
typedef __attribute__((ext_vector_type(8)))  float    v8f;

#define NVOX    32768          // 32^3
#define BN_INV  0.99950037f    // 1/sqrt(1 + 1e-3)

__device__ __forceinline__ int clampi(int v, int lo, int hi) {
    return v < lo ? lo : (v > hi ? hi : v);
}

// ---------------------------------------------------------------------------
// Dense GEMM: out[M x Nc] = epilogue(A[M x Kpad](f16) @ Wt[Ncpad x Kpad](f16)^T)
// One wave per 16x16 tile. mode==1: out = g*relu(acc+b)*BN_INV + bb.
// ---------------------------------------------------------------------------
__global__ void __launch_bounds__(32)
dense_wmma_kernel(const _Float16* __restrict__ A, const _Float16* __restrict__ Wt,
                  const float* __restrict__ bias, const float* __restrict__ gamma,
                  const float* __restrict__ beta, float* __restrict__ out,
                  int M, int Kpad, int Nc, int mode)
{
    const int lane = threadIdx.x;
    const int hi   = lane >> 4;      // 0: lanes 0-15, 1: lanes 16-31
    const int lm   = lane & 15;
    const int row0 = blockIdx.x * 16;
    const int col0 = blockIdx.y * 16;
    const int col  = col0 + lm;

    const _Float16* __restrict__ arow = A  + (size_t)(row0 + lm) * Kpad;
    const _Float16* __restrict__ brow = Wt + (size_t)col * Kpad;

    v8f acc = {};
    for (int k0 = 0; k0 < Kpad; k0 += 32) {
        v16h a = *(const v16h*)(arow + k0 + hi * 16);       // K 0..15 / 16..31
        v16h b = *(const v16h*)(brow + k0 + hi * 16);       // K 0..15 / 16..31
        acc = __builtin_amdgcn_wmma_f32_16x16x32_f16(
            false, a, false, b, (short)0, acc, false, false);
    }

    if (col >= Nc) return;
    const float bv = bias[col];
    const float gv = (mode == 1) ? gamma[col] : 1.f;
    const float bb = (mode == 1) ? beta[col] : 0.f;
#pragma unroll
    for (int r = 0; r < 8; ++r) {
        int row = row0 + hi * 8 + r;
        float v = acc[r] + bv;
        if (mode == 1) v = gv * (fmaxf(v, 0.f) * BN_INV) + bb;
        out[(size_t)row * Nc + col] = v;
    }
}

// ---------------------------------------------------------------------------
// 3x3x3 conv over the 32^3 grid as implicit GEMM on packed f16 operands.
// voxh has NVOX+1 rows of dipad channels; row NVOX is all-zero (halo).
// Wct is [dco][27*dipad] so each (tap, chunk) B fragment is contiguous.
// ---------------------------------------------------------------------------
__global__ void __launch_bounds__(32)
conv_wmma_kernel(const _Float16* __restrict__ voxh, const _Float16* __restrict__ Wct,
                 const float* __restrict__ bc, const float* __restrict__ gc,
                 const float* __restrict__ bbc, float* __restrict__ outg,
                 int dipad, int dco)
{
    const int lane = threadIdx.x;
    const int hi   = lane >> 4;
    const int lm   = lane & 15;
    const int mt   = blockIdx.x;              // 2048 tiles of 16 voxels (z-major)
    const int col0 = blockIdx.y * 16;
    const int col  = col0 + lm;

    const int iz0 = (mt & 1) * 16;
    const int iy  = (mt >> 1) & 31;
    const int ix  = mt >> 6;
    const int iz  = iz0 + lm;                 // this lane's A-row voxel

    const int K27 = 27 * dipad;
    const _Float16* __restrict__ brow = Wct + (size_t)col * K27;

    v8f acc = {};
#pragma unroll 1
    for (int t = 0; t < 27; ++t) {
        const int dx = t / 9 - 1, dy = (t / 3) % 3 - 1, dz = t % 3 - 1;
        const int nx = ix + dx, ny = iy + dy, nz = iz + dz;
        const bool ok = ((unsigned)nx < 32u) & ((unsigned)ny < 32u) & ((unsigned)nz < 32u);
        const int nb = ok ? ((nx * 32 + ny) * 32 + nz) : NVOX;   // halo row if OOB
        const _Float16* __restrict__ arow = voxh + (size_t)nb * dipad;
        const _Float16* __restrict__ btap = brow + (size_t)t * dipad;
        for (int c0 = 0; c0 < dipad; c0 += 32) {
            v16h a = *(const v16h*)(arow + c0 + hi * 16);
            v16h b = *(const v16h*)(btap + c0 + hi * 16);
            acc = __builtin_amdgcn_wmma_f32_16x16x32_f16(
                false, a, false, b, (short)0, acc, false, false);
        }
    }

    const float bv = bc[col], gv = gc[col], bb = bbc[col];
#pragma unroll
    for (int r = 0; r < 8; ++r) {
        int z = iz0 + hi * 8 + r;
        size_t v = (size_t)((ix * 32 + iy) * 32 + z);
        float val = acc[r] + bv;
        val = gv * (fmaxf(val, 0.f) * BN_INV) + bb;
        outg[v * dco + col] = val;
    }
}

// ---------------------------------------------------------------------------
// Packing / elementwise kernels
// ---------------------------------------------------------------------------

// f32 [n][K] -> f16 [n][Kpad], zero-padded K.
__global__ void pack_act_kernel(const float* __restrict__ x, _Float16* __restrict__ xh,
                                long total, int K, int Kpad)
{
    long idx = (long)blockIdx.x * blockDim.x + threadIdx.x;
    if (idx >= total) return;
    int row = (int)(idx / Kpad), c = (int)(idx % Kpad);
    xh[idx] = (c < K) ? (_Float16)x[(size_t)row * K + c] : (_Float16)0.f;
}

// W f32 [K][Nc] -> Wt f16 [Ncpad][Kpad] (transposed, zero-padded).
__global__ void pack_wd_kernel(const float* __restrict__ W, _Float16* __restrict__ Wt,
                               long total, int K, int Kpad, int Nc)
{
    long idx = (long)blockIdx.x * blockDim.x + threadIdx.x;
    if (idx >= total) return;
    int col = (int)(idx / Kpad), k = (int)(idx % Kpad);
    Wt[idx] = (k < K && col < Nc) ? (_Float16)W[(size_t)k * Nc + col] : (_Float16)0.f;
}

// Wc f32 [27][di][dco] -> Wct f16 [dco][27*dipad].
__global__ void pack_wc_kernel(const float* __restrict__ Wc, _Float16* __restrict__ Wct,
                               long total, int di, int dipad, int dco)
{
    long idx = (long)blockIdx.x * blockDim.x + threadIdx.x;
    if (idx >= total) return;
    int K27 = 27 * dipad;
    int col = (int)(idx / K27), r = (int)(idx % K27);
    int t = r / dipad, c = r % dipad;
    Wct[idx] = (c < di) ? (_Float16)Wc[((size_t)t * di + c) * dco + col] : (_Float16)0.f;
}

// vox f32 [NVOX][di] * ppv -> voxh f16 [NVOX+1][dipad]; row NVOX = zeros (halo).
__global__ void scale_cvt_kernel(const float* __restrict__ vox, const float* __restrict__ ppv,
                                 _Float16* __restrict__ voxh, long total, int di, int dipad)
{
    long idx = (long)blockIdx.x * blockDim.x + threadIdx.x;
    if (idx >= total) return;
    int v = (int)(idx / dipad), c = (int)(idx % dipad);
    float val = (v < NVOX && c < di) ? vox[(size_t)v * di + c] * ppv[v] : 0.f;
    voxh[idx] = (_Float16)val;
}

// ---------------------------------------------------------------------------
// Voxelization / scatter / devox / reductions
// ---------------------------------------------------------------------------
__global__ void voxelize_kernel(const float* __restrict__ pts, float* __restrict__ fc,
                                int* __restrict__ vid, int* __restrict__ cnt, int n)
{
    int i = blockIdx.x * blockDim.x + threadIdx.x;
    if (i >= n) return;
    float fx = pts[i * 3 + 0] * 32.f;
    float fy = pts[i * 3 + 1] * 32.f;
    float fz = pts[i * 3 + 2] * 32.f;
    fc[i * 3 + 0] = fx; fc[i * 3 + 1] = fy; fc[i * 3 + 2] = fz;
    int ix = clampi((int)floorf(fx), 0, 31);
    int iy = clampi((int)floorf(fy), 0, 31);
    int iz = clampi((int)floorf(fz), 0, 31);
    int v = (ix * 32 + iy) * 32 + iz;
    vid[i] = v;
    atomicAdd(&cnt[v], 1);
}

__global__ void ppv_kernel(const int* __restrict__ cnt, float* __restrict__ ppv)
{
    int v = blockIdx.x * blockDim.x + threadIdx.x;
    if (v >= NVOX) return;
    int c = cnt[v];
    ppv[v] = (c > 0) ? 1.0f / (float)c : 0.f;
}

__global__ void scatter_kernel(const float* __restrict__ x, const int* __restrict__ vid,
                               float* __restrict__ vox, long total, int C)
{
    long idx = (long)blockIdx.x * blockDim.x + threadIdx.x;
    if (idx >= total) return;
    int i = (int)(idx / C), c = (int)(idx % C);
    atomicAdd(&vox[(size_t)vid[i] * C + c], x[idx]);
}

__global__ void trilinear_kernel(const float* __restrict__ cg, const float* __restrict__ fc,
                                 float* __restrict__ out, long total, int C)
{
    long idx = (long)blockIdx.x * blockDim.x + threadIdx.x;
    if (idx >= total) return;
    int i = (int)(idx / C), c = (int)(idx % C);
    float fx = fc[i * 3 + 0], fy = fc[i * 3 + 1], fz = fc[i * 3 + 2];
    int x0 = (int)floorf(fx), y0 = (int)floorf(fy), z0 = (int)floorf(fz);
    float dx = fx - (float)x0, dy = fy - (float)y0, dz = fz - (float)z0;
    int x0c = clampi(x0, 0, 31), x1c = clampi(x0 + 1, 0, 31);
    int y0c = clampi(y0, 0, 31), y1c = clampi(y0 + 1, 0, 31);
    int z0c = clampi(z0, 0, 31), z1c = clampi(z0 + 1, 0, 31);
    float s = 0.f;
#pragma unroll
    for (int a = 0; a < 2; ++a)
#pragma unroll
        for (int b = 0; b < 2; ++b)
#pragma unroll
            for (int d = 0; d < 2; ++d) {
                int xi = a ? x1c : x0c, yi = b ? y1c : y0c, zi = d ? z1c : z0c;
                float w = (a ? dx : 1.f - dx) * (b ? dy : 1.f - dy) * (d ? dz : 1.f - dz);
                s += w * cg[((size_t)((xi * 32 + yi) * 32 + zi)) * C + c];
            }
    out[idx] += s;
}

__global__ void gmax_kernel(const float* __restrict__ x, float* __restrict__ g,
                            int n, int C)
{
    int c = blockIdx.x * blockDim.x + threadIdx.x;
    if (c >= C) return;
    float m = -3.4e38f;
    for (int i = 0; i < n; ++i) m = fmaxf(m, x[(size_t)i * C + c]);
    g[c] = m;
}

__global__ void concat_kernel(const float* __restrict__ loc, const float* __restrict__ g,
                              float* __restrict__ out, long total)
{
    long idx = (long)blockIdx.x * blockDim.x + threadIdx.x;
    if (idx >= total) return;
    int i = (int)(idx / 544), c = (int)(idx % 544);
    out[idx] = (c < 32) ? loc[(size_t)i * 32 + c] : g[c - 32];
}

// ---------------------------------------------------------------------------
// Host-side orchestration
// ---------------------------------------------------------------------------
static inline int round32(int v) { return (v + 31) & ~31; }
static inline long cdiv(long a, long b) { return (a + b - 1) / b; }

static void run_pvconv_layer(const float* xin, int di, int dco,
                             const float* Wm, const float* bm, const float* gm, const float* bbm,
                             const float* Wc, const float* bc, const float* gc, const float* bbc,
                             float* xout, float* vox, float* cg,
                             _Float16* xh, _Float16* voxh, _Float16* Wmt, _Float16* Wct,
                             const int* vid, const float* ppv, const float* fc,
                             int n, hipStream_t stream)
{
    const int dipad = round32(di);

    // ---- point branch: pack + WMMA GEMM with fused bias/ReLU/BN ----
    long tp = (long)n * dipad;
    pack_act_kernel<<<(int)cdiv(tp, 256), 256, 0, stream>>>(xin, xh, tp, di, dipad);
    long tw = (long)dco * dipad;
    pack_wd_kernel<<<(int)cdiv(tw, 256), 256, 0, stream>>>(Wm, Wmt, tw, di, dipad, dco);
    dim3 gdense(n / 16, dco / 16);
    dense_wmma_kernel<<<gdense, 32, 0, stream>>>(xh, Wmt, bm, gm, bbm, xout, n, dipad, dco, 1);

    // ---- voxel branch: scatter-mean -> f16 grid (+halo) -> WMMA conv -> devox ----
    hipMemsetAsync(vox, 0, (size_t)NVOX * di * sizeof(float), stream);
    long tsc = (long)n * di;
    scatter_kernel<<<(int)cdiv(tsc, 256), 256, 0, stream>>>(xin, vid, vox, tsc, di);
    long tvx = (long)(NVOX + 1) * dipad;
    scale_cvt_kernel<<<(int)cdiv(tvx, 256), 256, 0, stream>>>(vox, ppv, voxh, tvx, di, dipad);
    long twc = (long)dco * 27 * dipad;
    pack_wc_kernel<<<(int)cdiv(twc, 256), 256, 0, stream>>>(Wc, Wct, twc, di, dipad, dco);
    conv_wmma_kernel<<<dim3(2048, dco / 16), 32, 0, stream>>>(voxh, Wct, bc, gc, bbc, cg, dipad, dco);
    long tt = (long)n * dco;
    trilinear_kernel<<<(int)cdiv(tt, 256), 256, 0, stream>>>(cg, fc, xout, tt, dco);
}

extern "C" void kernel_launch(void* const* d_in, const int* in_sizes, int n_in,
                              void* d_out, int out_size, void* d_ws, size_t ws_size,
                              hipStream_t stream)
{
    (void)in_sizes; (void)n_in; (void)out_size; (void)ws_size;
    const int N = 100000;

    const float* pts = (const float*)d_in[0];
    auto P = [&](int layer, int slot) { return (const float*)d_in[1 + layer * 8 + slot]; };
    const float* Wf = (const float*)d_in[49];
    const float* bf = (const float*)d_in[50];

    // Workspace layout (in float units), 256B-aligned chunks.
    auto align64 = [](size_t v) { return (v + 63) & ~(size_t)63; };
    float* ws = (float*)d_ws;
    size_t o = 0;
    float* fc   = ws + o;           o += align64((size_t)N * 3);
    int*   vid  = (int*)(ws + o);   o += align64((size_t)N);
    int*   cnt  = (int*)(ws + o);   o += align64((size_t)NVOX);
    float* ppv  = ws + o;           o += align64((size_t)NVOX);
    float* vox  = ws + o;           o += align64((size_t)NVOX * 544);
    float* cg   = ws + o;           o += align64((size_t)NVOX * 512);
    float* bufA = ws + o;           o += align64((size_t)N * 544);
    float* bufB = ws + o;           o += align64((size_t)N * 544);
    float* loc  = ws + o;           o += align64((size_t)N * 32);
    float* gvec = ws + o;           o += align64((size_t)512);
    _Float16* xh   = (_Float16*)(ws + o); o += align64((size_t)N * 544 / 2);
    _Float16* voxh = (_Float16*)(ws + o); o += align64((size_t)(NVOX + 1) * 544 / 2 + 64);
    _Float16* Wmt  = (_Float16*)(ws + o); o += align64((size_t)160000 / 2);
    _Float16* Wct  = (_Float16*)(ws + o); o += align64((size_t)4200000 / 2);

    // Voxelization (shared by all layers)
    hipMemsetAsync(cnt, 0, NVOX * sizeof(int), stream);
    voxelize_kernel<<<(N + 255) / 256, 256, 0, stream>>>(pts, fc, vid, cnt, N);
    ppv_kernel<<<(NVOX + 255) / 256, 256, 0, stream>>>(cnt, ppv);

    // l0..l2
    run_pvconv_layer(pts, 3, 32, P(0,0),P(0,1),P(0,2),P(0,3),P(0,4),P(0,5),P(0,6),P(0,7),
                     loc, vox, cg, xh, voxh, Wmt, Wct, vid, ppv, fc, N, stream);
    run_pvconv_layer(loc, 32, 64, P(1,0),P(1,1),P(1,2),P(1,3),P(1,4),P(1,5),P(1,6),P(1,7),
                     bufA, vox, cg, xh, voxh, Wmt, Wct, vid, ppv, fc, N, stream);
    run_pvconv_layer(bufA, 64, 512, P(2,0),P(2,1),P(2,2),P(2,3),P(2,4),P(2,5),P(2,6),P(2,7),
                     bufB, vox, cg, xh, voxh, Wmt, Wct, vid, ppv, fc, N, stream);

    // global max + concat -> bufA (544)
    gmax_kernel<<<2, 256, 0, stream>>>(bufB, gvec, N, 512);
    long tcat = (long)N * 544;
    concat_kernel<<<(int)cdiv(tcat, 256), 256, 0, stream>>>(loc, gvec, bufA, tcat);

    // l3..l5
    run_pvconv_layer(bufA, 544, 256, P(3,0),P(3,1),P(3,2),P(3,3),P(3,4),P(3,5),P(3,6),P(3,7),
                     bufB, vox, cg, xh, voxh, Wmt, Wct, vid, ppv, fc, N, stream);
    run_pvconv_layer(bufB, 256, 128, P(4,0),P(4,1),P(4,2),P(4,3),P(4,4),P(4,5),P(4,6),P(4,7),
                     bufA, vox, cg, xh, voxh, Wmt, Wct, vid, ppv, fc, N, stream);
    run_pvconv_layer(bufA, 128, 64, P(5,0),P(5,1),P(5,2),P(5,3),P(5,4),P(5,5),P(5,6),P(5,7),
                     bufB, vox, cg, xh, voxh, Wmt, Wct, vid, ppv, fc, N, stream);

    // classifier: (64 -> 23), Ncpad = 32 (zero-padded weight rows), plain bias.
    long tp = (long)N * 64;
    pack_act_kernel<<<(int)cdiv(tp, 256), 256, 0, stream>>>(bufB, xh, tp, 64, 64);
    long tw = (long)32 * 64;
    pack_wd_kernel<<<(int)cdiv(tw, 256), 256, 0, stream>>>(Wf, Wmt, tw, 64, 64, 23);
    float* out = (float*)d_out;
    dense_wmma_kernel<<<dim3(N / 16, 2), 32, 0, stream>>>(xh, Wmt, bf, bf, bf, out, N, 64, 23, 0);
}
